// Model_39676907887447
// MI455X (gfx1250) — matrix operand order repaired and checked by reading, not compiled, not run on hardware
//
#include <hip/hip_runtime.h>

// Flash-attention forward, fp32 in/out, f16 WMMA compute with f32 accumulate.
// B=16, S=2048, D=128. Scale = 1/sqrt(128), folded with log2(e) into Q staging.
//
// Double-buffered pipeline per 64-key block (32 iterations, ONE barrier each):
//   stage(regs[block j+1] -> LDS buf[1-par]) ;
//   prefetch(block j+2 -> regs)              ;
//   compute(block j from buf[par]) : 16 WMMA K.Q^T, online softmax, 16 WMMA P.V
//   barrier
// Writes to a buffer are separated from its readers by a full barrier interval.

typedef __attribute__((ext_vector_type(16))) _Float16 v16h;
typedef __attribute__((ext_vector_type(8)))  _Float16 v8h;
typedef __attribute__((ext_vector_type(4)))  _Float16 v4h;
typedef __attribute__((ext_vector_type(2)))  _Float16 v2h;
typedef __attribute__((ext_vector_type(8)))  float    v8f;

#define S_LEN 2048
#define D_DIM 128
#define QTILE 128   // query rows per workgroup (8 waves x 16 rows)
#define KBLK  64    // key rows per inner block (four 16-key WMMA tiles)

union F16Frag { v16h v; v8h h[2]; };

__device__ __forceinline__ void load_block(const float* __restrict__ Kb,
                                           const float* __restrict__ Vb,
                                           int jb, int tid,
                                           float4* kr, float4* vr)
{
    #pragma unroll
    for (int u = 0; u < 8; ++u) {
        int i  = tid + u * 256;           // 0..2047 float4 units
        int r  = i >> 5;                  // 32 float4 per 128-elem row
        int c4 = i & 31;
        int off = (jb + r) * D_DIM + c4 * 4;   // 32-bit offsets -> saddr form
        kr[u] = *(const float4*)(Kb + off);
        vr[u] = *(const float4*)(Vb + off);
    }
}

__device__ __forceinline__ void stage_block(int tid,
                                            const float4* kr, const float4* vr,
                                            _Float16* Ks, _Float16* Vts)
{
    #pragma unroll
    for (int u = 0; u < 8; ++u) {
        int i  = tid + u * 256;
        int r  = i >> 5;
        int c4 = i & 31;
        v4h h = { (_Float16)kr[u].x, (_Float16)kr[u].y,
                  (_Float16)kr[u].z, (_Float16)kr[u].w };
        *(v4h*)(Ks + r * D_DIM + c4 * 4) = h;
        Vts[(c4 * 4 + 0) * KBLK + r] = (_Float16)vr[u].x;
        Vts[(c4 * 4 + 1) * KBLK + r] = (_Float16)vr[u].y;
        Vts[(c4 * 4 + 2) * KBLK + r] = (_Float16)vr[u].z;
        Vts[(c4 * 4 + 3) * KBLK + r] = (_Float16)vr[u].w;
    }
}

__global__ __launch_bounds__(256)
void fa_fwd_kernel(const float* __restrict__ Qg,
                   const float* __restrict__ Kg,
                   const float* __restrict__ Vg,
                   float* __restrict__ Og)
{
    // 80 KB LDS: two 32KB K/V^T buffers (double-buffered) + 16KB P tiles.
    // Phase-1 Q staging (32KB) reuses buffer 0 before the pipeline starts.
    __shared__ __align__(16) _Float16 lds[40960];

    const int  tid  = threadIdx.x;
    const int  wave = tid >> 5;
    const int  lane = tid & 31;
    const int  lh   = lane & 15;
    const bool lo   = lane < 16;

    const int b   = blockIdx.x >> 4;      // 16 q-blocks per batch
    const int qwg = (blockIdx.x & 15) * QTILE;
    const int qw  = qwg + wave * 16;      // first query row of this wave

    const float* Qb = Qg + (size_t)b * S_LEN * D_DIM;
    const float* Kb = Kg + (size_t)b * S_LEN * D_DIM;
    const float* Vb = Vg + (size_t)b * S_LEN * D_DIM;
    float*       Ob = Og + (size_t)b * S_LEN * D_DIM;

    // log2(e) / sqrt(128), folded into Q during staging
    const float kQScale = 1.442695040889f / 11.313708498985f;

    // issue block-0 K/V prefetch first; drains behind Q staging
    float4 kr[8], vr[8];
    load_block(Kb, Vb, 0, tid, kr, vr);

    // ---- Phase 1: stage workgroup's Q tile (pre-scaled) into LDS as f16 ----
    _Float16* Qs = lds;   // [128][128], overlaps buffer 0
    for (int i = tid; i < QTILE * (D_DIM / 4); i += 256) {
        int r  = i >> 5;
        int c4 = i & 31;
        float4 f = *(const float4*)(Qb + (qwg + r) * D_DIM + c4 * 4);
        v4h h = { (_Float16)(f.x * kQScale), (_Float16)(f.y * kQScale),
                  (_Float16)(f.z * kQScale), (_Float16)(f.w * kQScale) };
        *(v4h*)(Qs + r * D_DIM + c4 * 4) = h;
    }
    __syncthreads();

    // ---- Build Q^T B-fragments (4 chunks of 32 along D), kept in VGPRs ----
    v16h qt[4];
    {
        const int r = wave * 16 + lh;     // this lane's query row (B column)
        #pragma unroll
        for (int c = 0; c < 4; ++c) {
            F16Frag f;
            const v8h* p = (const v8h*)(Qs + r * D_DIM + c * 32 + (lo ? 0 : 16));
            f.h[0] = p[0];
            f.h[1] = p[1];
            qt[c] = f.v;
        }
    }
    __syncthreads();   // all waves done reading Q region (buffer 0 reusable)

    // ---- Pipeline prologue: stage block 0 into buffer 0, prefetch block 1 ----
    stage_block(tid, kr, vr, lds, lds + 8192);
    load_block(Kb, Vb, KBLK, tid, kr, vr);
    __syncthreads();   // buffer 0 visible

    // Output accumulators: 8 d-tiles of 16x16 f32 (rows=queries, cols=d).
    v8f o[8];
    #pragma unroll
    for (int t = 0; t < 8; ++t) o[t] = (v8f){};
    float m_run = -1e30f;   // running row max (base-2 logit domain)
    float l_run = 0.0f;     // running row sum

    _Float16* Ps = lds + 32768 + wave * 1024;  // [16][64] per-wave P tile

    for (int jb = 0; jb < S_LEN; jb += KBLK) {
        const int par = (jb >> 6) & 1;
        _Float16* KsC = lds + par * 16384;         // current block (compute)
        _Float16* VtC = KsC + 8192;
        _Float16* KsN = lds + (par ^ 1) * 16384;   // next block (stage)
        _Float16* VtN = KsN + 8192;

        // stage block jb+KBLK (regs loaded one full iteration ago)
        stage_block(tid, kr, vr, KsN, VtN);
        // prefetch block jb+2*KBLK; drains behind this iteration's compute
        load_block(Kb, Vb, (jb + 2 * KBLK) & (S_LEN - 1), tid, kr, vr);

        // ---- S^T = K * Q^T : four 16(keys)x16(queries) tiles ----
        v8f st[4];
        #pragma unroll
        for (int g = 0; g < 4; ++g) st[g] = (v8f){};
        #pragma unroll
        for (int c = 0; c < 4; ++c) {
            #pragma unroll
            for (int g = 0; g < 4; ++g) {
                F16Frag a;
                const v8h* p = (const v8h*)(KsC + (g * 16 + lh) * D_DIM
                                                + c * 32 + (lo ? 0 : 16));
                a.h[0] = p[0]; a.h[1] = p[1];
                st[g] = __builtin_amdgcn_wmma_f32_16x16x32_f16(
                            false, a.v, false, qt[c], (short)0, st[g], false, false);
            }
        }

        // ---- online softmax over 64 keys (logits already in base-2 domain) ----
        float bmax = -1e30f;
        #pragma unroll
        for (int g = 0; g < 4; ++g)
            #pragma unroll
            for (int v = 0; v < 8; ++v)
                bmax = fmaxf(bmax, st[g][v]);
        bmax = fmaxf(bmax, __shfl_xor(bmax, 16));       // single cross-lane step
        float mnew = fmaxf(m_run, bmax);

        // rescale l and O only when the running max actually grew (wave-uniform)
        if (__any(mnew > m_run)) {
            float corr = exp2f(m_run - mnew);
            l_run *= corr;
            #pragma unroll
            for (int v = 0; v < 8; ++v) {
                float rs = __shfl(corr, lo ? v : (v + 8));
                #pragma unroll
                for (int t = 0; t < 8; ++t) o[t][v] *= rs;
            }
        }
        m_run = mnew;

        float psum = 0.0f;
        #pragma unroll
        for (int g = 0; g < 4; ++g)
            #pragma unroll
            for (int v = 0; v < 8; ++v) {
                st[g][v] = exp2f(st[g][v] - mnew);
                psum += st[g][v];
            }
        psum += __shfl_xor(psum, 16);
        l_run += psum;

        // ---- P^T (C layout) -> per-wave LDS tile P[q][k], packed f16 pairs ----
        #pragma unroll
        for (int g = 0; g < 4; ++g)
            #pragma unroll
            for (int v = 0; v < 8; v += 2) {
                v2h pk = { (_Float16)st[g][v], (_Float16)st[g][v + 1] };
                *(v2h*)(Ps + lh * KBLK + g * 16 + (lo ? 0 : 8) + v) = pk;
            }
        // Same-wave LDS RAW: DS ops are in-order per wave; wait data counter.
        asm volatile("s_wait_dscnt 0" ::: "memory");

        // Read P back as two WMMA A-fragments (16 queries x 32 keys each)
        F16Frag pa0, pa1;
        const v8h* pp = (const v8h*)(Ps + lh * KBLK + (lo ? 0 : 16));
        pa0.h[0] = pp[0]; pa0.h[1] = pp[1];
        pa1.h[0] = pp[4]; pa1.h[1] = pp[5];

        // ---- O += P * V  (B-fragments from transposed V: contiguous reads) ----
        #pragma unroll
        for (int t = 0; t < 8; ++t) {
            F16Frag vb0, vb1;
            const v8h* vp = (const v8h*)(VtC + (t * 16 + lh) * KBLK + (lo ? 0 : 16));
            vb0.h[0] = vp[0]; vb0.h[1] = vp[1];
            vb1.h[0] = vp[4]; vb1.h[1] = vp[5];
            o[t] = __builtin_amdgcn_wmma_f32_16x16x32_f16(
                       false, pa0.v, false, vb0.v, (short)0, o[t], false, false);
            o[t] = __builtin_amdgcn_wmma_f32_16x16x32_f16(
                       false, pa1.v, false, vb1.v, (short)0, o[t], false, false);
        }

        __syncthreads();   // single barrier: next iter may overwrite buf[par]
    }

    // ---- finalize: O /= l, write fp32 ----
    float linv = 1.0f / l_run;
    #pragma unroll
    for (int v = 0; v < 8; ++v) {
        float rs  = __shfl(linv, lo ? v : (v + 8));
        int   row = qw + v + (lo ? 0 : 8);
        #pragma unroll
        for (int t = 0; t < 8; ++t) {
            Ob[row * D_DIM + t * 16 + lh] = o[t][v] * rs;
        }
    }
}

extern "C" void kernel_launch(void* const* d_in, const int* in_sizes, int n_in,
                              void* d_out, int out_size, void* d_ws, size_t ws_size,
                              hipStream_t stream) {
    const float* Q = (const float*)d_in[0];
    const float* K = (const float*)d_in[1];
    const float* V = (const float*)d_in[2];
    float*       O = (float*)d_out;
    // grid = B * (S / QTILE) = 16 * 16
    dim3 grid(16 * (S_LEN / QTILE));
    fa_fwd_kernel<<<grid, 256, 0, stream>>>(Q, K, V, O);
}
